// TypeAwareHGATLayer_42056319763058
// MI455X (gfx1250) — matrix operand order repaired and checked by reading, not compiled, not run on hardware
//
#include <hip/hip_runtime.h>
#include <math.h>

// ---------------------------------------------------------------------------
// TypeAwareHGAT layer for MI455X (gfx1250, wave32, WMMA).
// Q/K/V projections via v_wmma_f32_16x16x32_bf16 (f32 accumulate).
// Weights pre-converted once into bf16 B-fragment layout; H/z tiles staged in
// LDS in bf16 A-fragment layout, so the GEMM hot loop is loads+WMMA only.
// ---------------------------------------------------------------------------

typedef __attribute__((ext_vector_type(16))) __bf16 v16bf;
typedef __attribute__((ext_vector_type(8)))  float  v8f;

#define NEGC (-1.0e30f)
#define INV_SQRT_D 0.08838834764831845f  // 1/sqrt(128)

// ordered-float atomic max (valid for smax initialized to a negative sentinel)
static __device__ __forceinline__ void atomicMaxF(float* a, float v) {
    if (v >= 0.0f) atomicMax((int*)a, __float_as_int(v));
    else           atomicMin((unsigned int*)a, __float_as_uint(v));
}

// ---------------------------------------------------------------------------
// Weight pre-conversion: f32 [128x128] row-major -> bf16 in B-fragment order.
// Fragment element e of lane(col,half) for (nt,kc) = W[n*128 + k] with
// n = nt*16+col, k = kc*32 + 16*half + e  (contiguous k order, same as the A staging).
// Linear layout index r = nt*128 + kc*32 + half*16 + col, elements contiguous.
// ---------------------------------------------------------------------------
__global__ void hgat_wprep(const float* __restrict__ WQ, const float* __restrict__ WK,
                           const float* __restrict__ WV,
                           __bf16* __restrict__ WQb, __bf16* __restrict__ WKb,
                           __bf16* __restrict__ WVb) {
    int g = blockIdx.x * blockDim.x + threadIdx.x;
    if (g >= 3 * 1024) return;
    int mat  = g >> 10;
    int r    = g & 1023;
    int col  = r & 15;
    int half = (r >> 4) & 1;
    int kc   = (r >> 5) & 3;
    int nt   = r >> 7;
    const float* W  = (mat == 0) ? WQ  : (mat == 1) ? WK  : WV;
    __bf16*      Wb = (mat == 0) ? WQb : (mat == 1) ? WKb : WVb;
    int n     = nt * 16 + col;
    int kbase = kc * 32 + 16 * half;
    #pragma unroll
    for (int e = 0; e < 16; ++e)
        Wb[r * 16 + e] = (__bf16)W[n * 128 + kbase + e];
}

// ---------------------------------------------------------------------------
// init: zero output + denom, smax = NEG sentinel
// ---------------------------------------------------------------------------
__global__ void hgat_init(float* __restrict__ out, float* __restrict__ smax,
                          float* __restrict__ denom, int n_nodes) {
    int t = blockIdx.x * blockDim.x + threadIdx.x;
    if (t < n_nodes) { smax[t] = NEGC; denom[t] = 0.0f; }
    int total = n_nodes * 128;
    if (t < total) out[t] = 0.0f;
}

// ---------------------------------------------------------------------------
// Stage one 8-float run of a row into LDS in A-fragment order (bf16).
// Source k (multiple of 8) maps to group g = kc*2 + half, octet e0:
//   r = k & 31, half = r>>4, e0 = 8*((r>>3)&1).
// Inverse of A 16x32 mapping k%32 = e + 16*half (matches B fragments).
// ---------------------------------------------------------------------------
#define SA_STRIDE 144  // 16 bf16 x 9 groups (1 pad group): 288B rows, spreads banks

static __device__ __forceinline__ void stage_octet(__bf16* rowp, int k, const float* h) {
    int kc   = k >> 5;
    int r    = k & 31;
    int half = r >> 4;
    int e0   = 8 * ((r >> 3) & 1);
    union { __bf16 b[8]; uint4 u; } pk;
    #pragma unroll
    for (int j = 0; j < 8; ++j) pk.b[j] = (__bf16)h[j];
    *(uint4*)&rowp[(kc * 2 + half) * 16 + e0] = pk.u;
}

// ---------------------------------------------------------------------------
// Q projection: Q[n] = z[n] @ WQ^T for the stacked [user; item] rows.
// One wave = one 16-row tile; 8 n-tiles x 4 k-chunks of WMMA.
// ---------------------------------------------------------------------------
__global__ __launch_bounds__(128)
void hgat_q(const float* __restrict__ zu, const float* __restrict__ zi,
            const __bf16* __restrict__ WQb, float* __restrict__ Q,
            int U_, int N_) {
    __shared__ __attribute__((aligned(32))) __bf16 sA[4][16][SA_STRIDE];
    const int w    = threadIdx.x >> 5;
    const int lane = threadIdx.x & 31;
    const int tile = blockIdx.x * 4 + w;
    const int base = tile * 16;

    if (base < N_) {
        const int nl = lane >> 1;
        const int ng = base + nl;
        const int dbase = (lane & 1) * 64;
        const bool valid = (ng < N_);
        const float* src = valid ? ((ng < U_) ? (zu + (long long)ng * 128)
                                              : (zi + (long long)(ng - U_) * 128))
                                 : zu;
        #pragma unroll
        for (int d = 0; d < 64; d += 8) {
            float h[8];
            if (valid) {
                float4 a = *(const float4*)(src + dbase + d);
                float4 b = *(const float4*)(src + dbase + d + 4);
                h[0] = a.x; h[1] = a.y; h[2] = a.z; h[3] = a.w;
                h[4] = b.x; h[5] = b.y; h[6] = b.z; h[7] = b.w;
            } else {
                #pragma unroll
                for (int j = 0; j < 8; ++j) h[j] = 0.0f;
            }
            stage_octet(&sA[w][nl][0], dbase + d, h);
        }
    }
    __syncthreads();
    if (base >= N_) return;

    const int row  = lane & 15;   // A: M row / B,D: N col
    const int half = lane >> 4;

    v8f acc[8];
    #pragma unroll
    for (int nt = 0; nt < 8; ++nt) acc[nt] = (v8f){};

    #pragma unroll
    for (int kc = 0; kc < 4; ++kc) {
        v16bf a = *(const v16bf*)&sA[w][row][(kc * 2 + half) * 16];
        #pragma unroll
        for (int nt = 0; nt < 8; ++nt) {
            v16bf b = *(const v16bf*)(WQb + (nt * 128 + kc * 32 + half * 16 + row) * 16);
            acc[nt] = __builtin_amdgcn_wmma_f32_16x16x32_bf16(
                false, a, false, b, (short)0, acc[nt], false, false);
        }
    }

    #pragma unroll
    for (int nt = 0; nt < 8; ++nt) {
        #pragma unroll
        for (int r = 0; r < 8; ++r) {
            int mloc = r + 8 * half;
            int ng2  = base + mloc;
            if (ng2 < N_) Q[(long long)ng2 * 128 + nt * 16 + row] = acc[nt][r];
        }
    }
}

// ---------------------------------------------------------------------------
// Fused H -> (K,V) projection. One wave = 16 edges.
// H = (z_user[u1] + z_item[i1] + second)/3 staged in LDS as bf16 fragments
// (never hits HBM), then K = H@WK^T + type_embed, V = H@WV^T via WMMA.
// ---------------------------------------------------------------------------
__global__ __launch_bounds__(128)
void hgat_kv(const float* __restrict__ zu, const float* __restrict__ zi,
             const __bf16* __restrict__ WKb, const __bf16* __restrict__ WVb,
             const float* __restrict__ TE, const int* __restrict__ et,
             const int* __restrict__ u1, const int* __restrict__ u2,
             const int* __restrict__ i1, const int* __restrict__ i2,
             float* __restrict__ Kh, float* __restrict__ Vh, int M_) {
    __shared__ __attribute__((aligned(32))) __bf16 sA[4][16][SA_STRIDE];
    __shared__ int sET[4][16];
    const int w    = threadIdx.x >> 5;
    const int lane = threadIdx.x & 31;
    const int tile = blockIdx.x * 4 + w;
    const int e_base = tile * 16;

    if (e_base < M_) {
        const int el = lane >> 1;
        const int eg = e_base + el;
        const int dbase = (lane & 1) * 64;
        const bool valid = (eg < M_);
        const float *pu = zu, *pi = zu, *p2 = zu;
        if (valid) {
            const int t = et[eg];
            pu = zu + (long long)u1[eg] * 128;
            pi = zi + (long long)i1[eg] * 128;
            p2 = (t == 0) ? (zi + (long long)i2[eg] * 128)
                          : (zu + (long long)u2[eg] * 128);
            if ((lane & 1) == 0) sET[w][el] = t;
        } else if ((lane & 1) == 0) {
            sET[w][el] = 0;
        }
        #pragma unroll
        for (int d = 0; d < 64; d += 8) {
            float h[8];
            if (valid) {
                float4 a0 = *(const float4*)(pu + dbase + d);
                float4 b0 = *(const float4*)(pi + dbase + d);
                float4 c0 = *(const float4*)(p2 + dbase + d);
                float4 a1 = *(const float4*)(pu + dbase + d + 4);
                float4 b1 = *(const float4*)(pi + dbase + d + 4);
                float4 c1 = *(const float4*)(p2 + dbase + d + 4);
                h[0] = (a0.x + b0.x + c0.x) * (1.f / 3.f);
                h[1] = (a0.y + b0.y + c0.y) * (1.f / 3.f);
                h[2] = (a0.z + b0.z + c0.z) * (1.f / 3.f);
                h[3] = (a0.w + b0.w + c0.w) * (1.f / 3.f);
                h[4] = (a1.x + b1.x + c1.x) * (1.f / 3.f);
                h[5] = (a1.y + b1.y + c1.y) * (1.f / 3.f);
                h[6] = (a1.z + b1.z + c1.z) * (1.f / 3.f);
                h[7] = (a1.w + b1.w + c1.w) * (1.f / 3.f);
            } else {
                #pragma unroll
                for (int j = 0; j < 8; ++j) h[j] = 0.0f;
            }
            stage_octet(&sA[w][el][0], dbase + d, h);
        }
    }
    __syncthreads();
    if (e_base >= M_) return;

    const int row  = lane & 15;
    const int half = lane >> 4;

    v8f accK[8], accV[8];
    #pragma unroll
    for (int nt = 0; nt < 8; ++nt) { accK[nt] = (v8f){}; accV[nt] = (v8f){}; }

    #pragma unroll
    for (int kc = 0; kc < 4; ++kc) {
        v16bf a = *(const v16bf*)&sA[w][row][(kc * 2 + half) * 16];
        #pragma unroll
        for (int nt = 0; nt < 8; ++nt) {
            int boff = (nt * 128 + kc * 32 + half * 16 + row) * 16;
            v16bf bk = *(const v16bf*)(WKb + boff);
            v16bf bv = *(const v16bf*)(WVb + boff);
            accK[nt] = __builtin_amdgcn_wmma_f32_16x16x32_bf16(
                false, a, false, bk, (short)0, accK[nt], false, false);
            accV[nt] = __builtin_amdgcn_wmma_f32_16x16x32_bf16(
                false, a, false, bv, (short)0, accV[nt], false, false);
        }
    }

    #pragma unroll
    for (int nt = 0; nt < 8; ++nt) {
        #pragma unroll
        for (int r = 0; r < 8; ++r) {
            int mloc = r + 8 * half;
            int eg2  = e_base + mloc;
            if (eg2 < M_) {
                int n = nt * 16 + row;
                Kh[(long long)eg2 * 128 + n] = accK[nt][r] + TE[sET[w][mloc] * 128 + n];
                Vh[(long long)eg2 * 128 + n] = accV[nt][r];
            }
        }
    }
}

// ---------------------------------------------------------------------------
// Incidence decode helper (2*M incidences per side; side 0 = users, 1 = items)
// ---------------------------------------------------------------------------
static __device__ __forceinline__ void decode_incidence(
    int side, int p, int M_, int U_,
    const int* u1, const int* u2, const int* i1, const int* i2, const int* et,
    int& m, int& node, bool& mask) {
    int slot2 = (p >= M_);
    m = slot2 ? (p - M_) : p;
    int seg;
    if (side == 0) { seg = slot2 ? u2[m] : u1[m]; mask = !slot2 || (et[m] == 1); node = seg; }
    else           { seg = slot2 ? i2[m] : i1[m]; mask = !slot2 || (et[m] == 0); node = U_ + seg; }
}

// ---------------------------------------------------------------------------
// Pass A: s = <Q[node], K[edge]> / sqrt(D); wave per incidence; atomic max.
// ---------------------------------------------------------------------------
__global__ __launch_bounds__(256)
void hgat_score(const float* __restrict__ Q, const float* __restrict__ Kh,
                const int* __restrict__ u1, const int* __restrict__ u2,
                const int* __restrict__ i1, const int* __restrict__ i2,
                const int* __restrict__ et,
                float* __restrict__ sU, float* __restrict__ sI,
                float* __restrict__ smax, int U_, int M_) {
    long long gw = ((long long)blockIdx.x * blockDim.x + threadIdx.x) >> 5;
    int lane = threadIdx.x & 31;
    long long P = 2LL * M_;
    if (gw >= 2 * P) return;
    int side = (gw >= P) ? 1 : 0;
    int p = (int)(gw - (long long)side * P);

    int m, node; bool mask;
    decode_incidence(side, p, M_, U_, u1, u2, i1, i2, et, m, node, mask);
    float* sArr = side ? sI : sU;
    if (!mask) { if (lane == 0) sArr[p] = NEGC; return; }

    const float4* q4 = (const float4*)(Q  + (long long)node * 128);
    const float4* k4 = (const float4*)(Kh + (long long)m * 128);
    float4 qa = q4[lane];
    float4 ka = k4[lane];
    float part = qa.x * ka.x + qa.y * ka.y + qa.z * ka.z + qa.w * ka.w;
    #pragma unroll
    for (int off = 16; off > 0; off >>= 1) part += __shfl_xor(part, off, 32);
    float sv = part * INV_SQRT_D;
    if (lane == 0) { sArr[p] = sv; atomicMaxF(&smax[node], sv); }
}

// ---------------------------------------------------------------------------
// Pass B: e = exp(s - smax); accumulate denom. Thread per incidence.
// ---------------------------------------------------------------------------
__global__ __launch_bounds__(256)
void hgat_exp(const int* __restrict__ u1, const int* __restrict__ u2,
              const int* __restrict__ i1, const int* __restrict__ i2,
              const int* __restrict__ et,
              float* __restrict__ sU, float* __restrict__ sI,
              const float* __restrict__ smax, float* __restrict__ denom,
              int U_, int M_) {
    long long t = (long long)blockIdx.x * blockDim.x + threadIdx.x;
    long long P = 2LL * M_;
    if (t >= 2 * P) return;
    int side = (t >= P) ? 1 : 0;
    int p = (int)(t - (long long)side * P);

    int m, node; bool mask;
    decode_incidence(side, p, M_, U_, u1, u2, i1, i2, et, m, node, mask);
    float* sArr = side ? sI : sU;
    float e = 0.0f;
    if (mask) e = expf(sArr[p] - smax[node]);
    sArr[p] = e;                 // overwrite score with exp value
    if (e > 0.0f) atomicAdd(&denom[node], e);
}

// ---------------------------------------------------------------------------
// Pass C: out[node] += (e/denom) * V[edge]; wave per incidence, f32 atomics.
// ---------------------------------------------------------------------------
__global__ __launch_bounds__(256)
void hgat_scatter(const float* __restrict__ Vh,
                  const float* __restrict__ eU, const float* __restrict__ eI,
                  const float* __restrict__ denom,
                  const int* __restrict__ u1, const int* __restrict__ u2,
                  const int* __restrict__ i1, const int* __restrict__ i2,
                  const int* __restrict__ et,
                  float* __restrict__ out, int U_, int M_) {
    long long gw = ((long long)blockIdx.x * blockDim.x + threadIdx.x) >> 5;
    int lane = threadIdx.x & 31;
    long long P = 2LL * M_;
    if (gw >= 2 * P) return;
    int side = (gw >= P) ? 1 : 0;
    int p = (int)(gw - (long long)side * P);

    int m, node; bool mask;
    decode_incidence(side, p, M_, U_, u1, u2, i1, i2, et, m, node, mask);
    const float* eArr = side ? eI : eU;
    float e = eArr[p];
    if (e == 0.0f) return;       // masked or zero weight: nothing to add
    float den = denom[node];
    float att = e / ((den == 0.0f) ? 1.0f : den);

    const float4* v4 = (const float4*)(Vh + (long long)m * 128);
    float4 v = v4[lane];
    float* o = out + (long long)node * 128 + lane * 4;
    atomicAdd(o + 0, att * v.x);
    atomicAdd(o + 1, att * v.y);
    atomicAdd(o + 2, att * v.z);
    atomicAdd(o + 3, att * v.w);
}

// ---------------------------------------------------------------------------
// launcher
// ---------------------------------------------------------------------------
extern "C" void kernel_launch(void* const* d_in, const int* in_sizes, int n_in,
                              void* d_out, int out_size, void* d_ws, size_t ws_size,
                              hipStream_t stream) {
    const float* zu = (const float*)d_in[0];
    const float* zi = (const float*)d_in[1];
    const float* WQ = (const float*)d_in[2];
    const float* WK = (const float*)d_in[3];
    const float* WV = (const float*)d_in[4];
    const float* TE = (const float*)d_in[5];
    const int* et = (const int*)d_in[6];
    const int* u1 = (const int*)d_in[7];
    const int* u2 = (const int*)d_in[8];
    const int* i1 = (const int*)d_in[9];
    const int* i2 = (const int*)d_in[10];

    const int U_ = in_sizes[0] / 128;
    const int I_ = in_sizes[1] / 128;
    const int M_ = in_sizes[6];
    const int N_ = U_ + I_;

    // workspace layout: bf16 weight fragments first (32B-aligned at base),
    // then the float region.
    __bf16* WQb = (__bf16*)d_ws;
    __bf16* WKb = WQb + 16384;
    __bf16* WVb = WKb + 16384;
    float* ws = (float*)(WVb + 16384);
    size_t o = 0;
    float* Q    = ws + o; o += (size_t)N_ * 128;
    float* Kh   = ws + o; o += (size_t)M_ * 128;
    float* Vh   = ws + o; o += (size_t)M_ * 128;
    float* sU   = ws + o; o += 2 * (size_t)M_;
    float* sI   = ws + o; o += 2 * (size_t)M_;
    float* smax = ws + o; o += (size_t)N_;
    float* denom= ws + o;

    float* out = (float*)d_out;

    hgat_wprep<<<(3 * 1024 + 255) / 256, 256, 0, stream>>>(WQ, WK, WV, WQb, WKb, WVb);

    hgat_init<<<(N_ * 128 + 255) / 256, 256, 0, stream>>>(out, smax, denom, N_);

    int qTiles = (N_ + 15) / 16;
    hgat_q<<<(qTiles + 3) / 4, 128, 0, stream>>>(zu, zi, WQb, Q, U_, N_);

    int kvTiles = (M_ + 15) / 16;
    hgat_kv<<<(kvTiles + 3) / 4, 128, 0, stream>>>(zu, zi, WKb, WVb, TE, et,
                                                   u1, u2, i1, i2, Kh, Vh, M_);

    long long nInc = 4LL * M_;                       // 2*M per side, 2 sides
    long long waveBlocks = (nInc * 32 + 255) / 256;  // wave per incidence
    hgat_score<<<(int)waveBlocks, 256, 0, stream>>>(Q, Kh, u1, u2, i1, i2, et,
                                                    sU, sI, smax, U_, M_);

    hgat_exp<<<(int)((nInc + 255) / 256), 256, 0, stream>>>(u1, u2, i1, i2, et,
                                                            sU, sI, smax, denom, U_, M_);

    hgat_scatter<<<(int)waveBlocks, 256, 0, stream>>>(Vh, sU, sI, denom,
                                                      u1, u2, i1, i2, et, out, U_, M_);
}
